// TransformerBlock3D_22110491640508
// MI455X (gfx1250) — matrix operand order repaired and checked by reading, not compiled, not run on hardware
//
#include <hip/hip_runtime.h>

typedef _Float16 h16;
typedef __attribute__((ext_vector_type(16))) _Float16 v16h;
typedef __attribute__((ext_vector_type(8)))  _Float16 v8h;
typedef __attribute__((ext_vector_type(8)))  float    v8f;
typedef __attribute__((ext_vector_type(4)))  unsigned int v4u;
typedef __attribute__((ext_vector_type(8)))  int v8i;
typedef __attribute__((ext_vector_type(4)))  int v4i;

#ifndef N_TOK
#define N_TOK 65536   // B=1, D=16, H=64, W=64
#endif
#define DIMC 192
#define SCALE_Q 0.17677669529663687f  // 32^-0.5

// Tensor Data Mover path.  This toolchain exposes the 6-arg builtin
// (g0 v4u, g1 v8i, g2 v4i, g3 v4i, v8i, cpol).
#if defined(__gfx1250__) && __has_builtin(__builtin_amdgcn_tensor_load_to_lds) && \
    __has_builtin(__builtin_amdgcn_s_wait_tensorcnt)
#define USE_TDM 1
#else
#define USE_TDM 0
#endif

// ---------------------------------------------------------------------------
// WMMA helpers (gfx1250 wave32, 16x16x32 f16 -> f32)
// ---------------------------------------------------------------------------
__device__ inline v8f wmma_f16(v16h a, v16h b, v8f c) {
  return __builtin_amdgcn_wmma_f32_16x16x32_f16(false, a, false, b, (short)0, c,
                                                false, false);
}

// A fragment: 16x32 (MxK), A row-major, lda multiple of 8.
// Contiguous-k layout used here (same as B): lanes 0-15 hold M=lane, halves 0..15 = K 0..15;
// lanes 16-31 hold K 16..31.  -> two contiguous 16B loads per lane.
__device__ inline v16h load_a16x32(const h16* a, int lda) {
  const int lane = threadIdx.x & 31;
  const int row = lane & 15;
  const int kb  = (lane >> 4) << 4;   // 0 or 16
  const h16* p = a + (size_t)row * lda + kb;
  v8h lo = *(const v8h*)(p);
  v8h hi = *(const v8h*)(p + 8);
  v16h f;
#pragma unroll
  for (int i = 0; i < 8; ++i) { f[i] = lo[i]; f[i + 8] = hi[i]; }
  return f;
}

// B fragment: 32x16 (KxN) from B^T stored N x K row-major (ldb multiple of 16).
// Contiguous-k layout used here (same as A): half i -> K = (lane/16)*16 + i, N = lane%16 -> one 32B load.
__device__ inline v16h load_bt16x32(const h16* bt, int ldb) {
  const int lane = threadIdx.x & 31;
  const int n  = lane & 15;
  const int kb = (lane >> 4) << 4;    // 0 or 16
  return *(const v16h*)(bt + (size_t)n * ldb + kb);
}

// 32(M) x 32(N) tile per wave: 2 A-frags x 2 B-frags -> 4 WMMAs per K step.
// Halves A/B panel re-reads vs a 16-wide tile (we are L2/HBM bound).
template <int K>
__device__ inline void gemm_tile_2x2(const h16* __restrict__ A,
                                     const h16* __restrict__ Bt,
                                     int m0, int n0,
                                     v8f& c00, v8f& c01, v8f& c10, v8f& c11) {
  const h16* a0 = A  + (size_t)m0 * K;
  const h16* a1 = a0 + (size_t)16 * K;
  const h16* b0 = Bt + (size_t)n0 * K;
  const h16* b1 = b0 + (size_t)16 * K;
#pragma unroll
  for (int k0 = 0; k0 < K; k0 += 32) {
    __builtin_prefetch(a0 + k0 + 32 * K, 0, 0);   // next M tile (speculative)
    v16h af0 = load_a16x32(a0 + k0, K);
    v16h af1 = load_a16x32(a1 + k0, K);
    v16h bf0 = load_bt16x32(b0 + k0, K);
    v16h bf1 = load_bt16x32(b1 + k0, K);
    c00 = wmma_f16(af0, bf0, c00);
    c01 = wmma_f16(af0, bf1, c01);
    c10 = wmma_f16(af1, bf0, c10);
    c11 = wmma_f16(af1, bf1, c11);
  }
}

__device__ inline float wave_sum(float v) {
#pragma unroll
  for (int m = 16; m >= 1; m >>= 1) v += __shfl_xor(v, m, 32);
  return v;
}

// ---------------------------------------------------------------------------
// Weight prep: f32 (K x N) -> f16 transposed (N x K); optional column scaling
// (folds attention 1/sqrt(hd) into the q-columns of qkv_w).
// ---------------------------------------------------------------------------
__global__ void k_convert_wT(const float* __restrict__ src, h16* __restrict__ dst,
                             int K, int N, float scale, int scale_cols) {
  int i = blockIdx.x * 256 + threadIdx.x;
  if (i >= K * N) return;
  int n = i / K, k = i - n * K;
  float v = src[(size_t)k * N + n];
  if (n < scale_cols) v *= scale;
  dst[(size_t)n * K + k] = (h16)v;
}

__global__ void k_scale_bias(const float* __restrict__ src, float* __restrict__ dst,
                             int n, int scale_cols, float scale) {
  int i = blockIdx.x * 256 + threadIdx.x;
  if (i < n) dst[i] = src[i] * (i < scale_cols ? scale : 1.0f);
}

// ---------------------------------------------------------------------------
// LayerNorm (192 ch) -> f16; optional window permutation of output rows.
// 4 waves / block, one token per wave.
// ---------------------------------------------------------------------------
__global__ __launch_bounds__(128)
void k_ln(const float* __restrict__ in, const float* __restrict__ gam,
          const float* __restrict__ bet, h16* __restrict__ out, int windowed) {
  const int wv = threadIdx.x >> 5, lane = threadIdx.x & 31;
  const int g = blockIdx.x * 4 + wv;
  const float* row = in + (size_t)g * DIMC;
  float v[6];
  float s = 0.f;
#pragma unroll
  for (int i = 0; i < 6; ++i) { v[i] = row[lane + 32 * i]; s += v[i]; }
  s = wave_sum(s);
  float mu = s * (1.f / DIMC);
  float q = 0.f;
#pragma unroll
  for (int i = 0; i < 6; ++i) { float d = v[i] - mu; q += d * d; }
  q = wave_sum(q);
  float rstd = rsqrtf(q * (1.f / DIMC) + 1e-5f);

  size_t t;
  if (windowed) {
    int gd = g >> 12, gh = (g >> 6) & 63, gw = g & 63;
    int wd = gd >> 2, d = gd & 3;
    int wh = gh >> 3, hh = gh & 7;
    int ww = gw >> 3, wl = gw & 7;
    int w = (wd * 8 + wh) * 8 + ww;
    int n = (d * 8 + hh) * 8 + wl;
    t = (size_t)w * 256 + n;
  } else {
    t = (size_t)g;
  }
  h16* orow = out + t * DIMC;
#pragma unroll
  for (int i = 0; i < 6; ++i) {
    int c = lane + 32 * i;
    orow[c] = (h16)((v[i] - mu) * rstd * gam[c] + bet[c]);
  }
}

// ---------------------------------------------------------------------------
// Generic GEMM: C_f16[M x N] = A_f16[M x K] @ Bt^T + bias.  One wave per
// 32x32 output tile.  grid = (M/32, N/32), block = 32.
// ---------------------------------------------------------------------------
template <int K>
__global__ __launch_bounds__(32)
void k_gemm_f16(const h16* __restrict__ A, const h16* __restrict__ Bt,
                const float* __restrict__ bias, h16* __restrict__ C, int N) {
  const int lane = threadIdx.x;
  const int m0 = blockIdx.x * 32;
  const int n0 = blockIdx.y * 32;
  v8f c00 = {}, c01 = {}, c10 = {}, c11 = {};
  gemm_tile_2x2<K>(A, Bt, m0, n0, c00, c01, c10, c11);
  const int col0 = n0 + (lane & 15), col1 = col0 + 16;
  const float bv0 = bias ? bias[col0] : 0.f;
  const float bv1 = bias ? bias[col1] : 0.f;
  const int mrow = m0 + ((lane >> 4) << 3);
#pragma unroll
  for (int r = 0; r < 8; ++r) {
    C[(size_t)(mrow + r) * N + col0]      = (h16)(c00[r] + bv0);
    C[(size_t)(mrow + r) * N + col1]      = (h16)(c01[r] + bv1);
    C[(size_t)(mrow + 16 + r) * N + col0] = (h16)(c10[r] + bv0);
    C[(size_t)(mrow + 16 + r) * N + col1] = (h16)(c11[r] + bv1);
  }
}

// ---------------------------------------------------------------------------
// Shifted-window attention.  One wave per (window, head, 16-row slab).
// qkv layout: [w*256+n][576] f16, q pre-scaled (cols 0..191), k (192..383),
// v (384..575).  The 256x32 K panel is DMA'd into LDS by the Tensor Data
// Mover (2D tile, stride 576), consumed after s_wait_tensorcnt.  S tile via
// 1 WMMA (K=hd=32); rel-bias + shift mask computed analytically; softmax in
// LDS; O = P@V via 8 WMMAs per 16-col tile.
// ---------------------------------------------------------------------------
__global__ __launch_bounds__(32)
void k_attn(const h16* __restrict__ qkv, h16* __restrict__ outp,
            const float* __restrict__ rel) {
  __shared__ alignas(32) float S[16][256];     // 16 KB
#if USE_TDM
  __shared__ alignas(64) h16   Klds[256][32];  // 16 KB (TDM destination)
#endif
  __shared__ alignas(32) h16   P[16][256];     //  8 KB
  __shared__ alignas(32) h16   Vt[32][256];    // 16 KB
  __shared__ int lab[256];                     //  1 KB

  const int lane = threadIdx.x;
  const int bx = blockIdx.x;
  const int w = bx / 96;
  const int r = bx - w * 96;
  const int head = r >> 4;
  const int slab = r & 15;
  const int m0 = slab * 16;
  const h16* base = qkv + (size_t)(w * 256) * 576;

#if USE_TDM
  {
    // D# for a 2D f16 tile: 32 (hd) x 256 (tokens), global row stride 576.
    unsigned long long ga = (unsigned long long)(base + 192 + head * 32);
    unsigned int lds_addr = (unsigned int)(size_t)(&Klds[0][0]);
    v4u g0;
    g0[0] = 1u;                                   // count=1, user mode
    g0[1] = lds_addr;                             // lds_addr [63:32]
    g0[2] = (unsigned int)(ga & 0xffffffffu);     // global_addr [95:64]
    g0[3] = (unsigned int)((ga >> 32) & 0x01ffffffu) | (2u << 30);  // type=2
    v8i g1;
    g1[0] = (int)(1u << 16);          // data_size = 1 (2 bytes)
    g1[1] = (int)(32u << 16);         // tensor_dim0 = 32
    g1[2] = (int)(256u << 16);        // tensor_dim1 = 256
    g1[3] = (int)(32u << 16);         // tile_dim0 = 32
    g1[4] = (int)256;                 // tile_dim1 = 256, tile_dim2 = 0
    g1[5] = (int)576;                 // tensor_dim0_stride = 576
    g1[6] = 0;
    g1[7] = 0;
    v4i z4 = {0, 0, 0, 0};
    v8i z8 = {0, 0, 0, 0, 0, 0, 0, 0};
    __builtin_amdgcn_tensor_load_to_lds(g0, g1, z4, z4, z8, 0);
  }
#endif

  // Stage V^T (lane = hd dim) and the shifted-window labels.
  for (int t = 0; t < 256; ++t)
    Vt[lane][t] = base[(size_t)t * 576 + 384 + head * 32 + lane];
  const int wd = w >> 6, wh = (w >> 3) & 7, ww = w & 7;
  for (int t = lane; t < 256; t += 32) {
    int d = t >> 6, hh = (t >> 3) & 7, wl = t & 7;
    int gd = wd * 4 + d, gh = wh * 8 + hh, gw = ww * 8 + wl;
    int sd = (gd + 2) & 15, sh = (gh + 4) & 63, sw = (gw + 4) & 63;  // roll by SS
    lab[t] = (sd >> 2) * 64 + (sh >> 3) * 8 + (sw >> 3);
  }
  __syncthreads();
#if USE_TDM
  __builtin_amdgcn_s_wait_tensorcnt(0);   // K panel resident in LDS
#endif

  // S = q @ k^T  (+ rel bias + mask)
  const v16h qf = load_a16x32(base + (size_t)m0 * 576 + head * 32, 576);
  const int Mloc = (lane >> 4) << 3;
  for (int n0 = 0; n0 < 256; n0 += 16) {
#if USE_TDM
    v16h kf = load_bt16x32(&Klds[n0][0], 32);
#else
    v16h kf = load_bt16x32(base + (size_t)n0 * 576 + 192 + head * 32, 576);
#endif
    v8f acc = {};
    acc = wmma_f16(qf, kf, acc);
    const int n = n0 + (lane & 15);
    const int kd = n >> 6, kh = (n >> 3) & 7, kw = n & 7;
    const int labn = lab[n];
#pragma unroll
    for (int rr = 0; rr < 8; ++rr) {
      int m = m0 + Mloc + rr;
      int qd = m >> 6, qh = (m >> 3) & 7, qw = m & 7;
      int idx = (qd - kd + 3) * 225 + (qh - kh + 7) * 15 + (qw - kw + 7);
      float b = rel[idx * 6 + head];
      float msk = (lab[m] == labn) ? 0.f : -100.f;
      S[Mloc + rr][n] = acc[rr] + b + msk;
    }
  }
  __syncthreads();

  // Row softmax (lanes 0..15 own one row each; 256 cols serial).
  if (lane < 16) {
    float mx = -1e30f;
    for (int j = 0; j < 256; ++j) mx = fmaxf(mx, S[lane][j]);
    float sum = 0.f;
    for (int j = 0; j < 256; ++j) {
      float e = __expf(S[lane][j] - mx);
      S[lane][j] = e;
      sum += e;
    }
    float rinv = 1.f / sum;
    for (int j = 0; j < 256; ++j) P[lane][j] = (h16)(S[lane][j] * rinv);
  }
  __syncthreads();

  // O = P(16x256) @ V(256x32)
  for (int nc = 0; nc < 2; ++nc) {
    v8f acc = {};
#pragma unroll
    for (int kt = 0; kt < 8; ++kt) {
      v16h pf = load_a16x32(&P[0][kt * 32], 256);
      v16h vf = load_bt16x32(&Vt[nc * 16][kt * 32], 256);
      acc = wmma_f16(pf, vf, acc);
    }
    const int col = head * 32 + nc * 16 + (lane & 15);
#pragma unroll
    for (int rr = 0; rr < 8; ++rr) {
      int tok = w * 256 + m0 + Mloc + rr;
      outp[(size_t)tok * DIMC + col] = (h16)acc[rr];
    }
  }
}

// ---------------------------------------------------------------------------
// Projection GEMM with fused window->global permutation and residual:
// x1[g] = x[g] + attn_out[t] @ proj_w + proj_b.  grid = (M/32, 192/32).
// ---------------------------------------------------------------------------
__global__ __launch_bounds__(32)
void k_proj_res(const h16* __restrict__ A, const h16* __restrict__ Bt,
                const float* __restrict__ bias, const float* __restrict__ xin,
                float* __restrict__ x1) {
  const int lane = threadIdx.x;
  const int m0 = blockIdx.x * 32, n0 = blockIdx.y * 32;
  v8f c00 = {}, c01 = {}, c10 = {}, c11 = {};
  gemm_tile_2x2<192>(A, Bt, m0, n0, c00, c01, c10, c11);
  const int col0 = n0 + (lane & 15), col1 = col0 + 16;
  const float bv0 = bias[col0], bv1 = bias[col1];
  const int Mloc = (lane >> 4) << 3;
#pragma unroll
  for (int mt = 0; mt < 2; ++mt) {
    v8f a0 = mt ? c10 : c00;
    v8f a1 = mt ? c11 : c01;
#pragma unroll
    for (int rr = 0; rr < 8; ++rr) {
      int t = m0 + mt * 16 + Mloc + rr;
      int wdw = t >> 8, n = t & 255;
      int wd = wdw >> 6, wh = (wdw >> 3) & 7, ww = wdw & 7;
      int d = n >> 6, hh = (n >> 3) & 7, wl = n & 7;
      int g = ((wd * 4 + d) << 12) + ((wh * 8 + hh) << 6) + (ww * 8 + wl);
      size_t o = (size_t)g * DIMC;
      x1[o + col0] = xin[o + col0] + a0[rr] + bv0;
      x1[o + col1] = xin[o + col1] + a1[rr] + bv1;
    }
  }
}

// ---------------------------------------------------------------------------
// Depthwise 3x3x3 conv + bias + exact GELU over [g][768] f16 activations.
// ---------------------------------------------------------------------------
__global__ __launch_bounds__(256)
void k_dwconv_gelu(const h16* __restrict__ in, const float* __restrict__ wgt,
                   const float* __restrict__ bias, h16* __restrict__ out) {
  const int g = blockIdx.x;
  const int gd = g >> 12, gh = (g >> 6) & 63, gw = g & 63;
  for (int c = threadIdx.x; c < 768; c += 256) {
    float acc = bias[c];
#pragma unroll
    for (int kd = 0; kd < 3; ++kd) {
      int pd = gd + kd - 1;
      if ((unsigned)pd >= 16u) continue;
#pragma unroll
      for (int kh = 0; kh < 3; ++kh) {
        int ph = gh + kh - 1;
        if ((unsigned)ph >= 64u) continue;
#pragma unroll
        for (int kw = 0; kw < 3; ++kw) {
          int pw = gw + kw - 1;
          if ((unsigned)pw >= 64u) continue;
          size_t gi = (size_t)(((pd << 6) + ph) << 6 | pw) * 768 + c;
          acc += (float)in[gi] * wgt[c * 27 + kd * 9 + kh * 3 + kw];
        }
      }
    }
    float ge = 0.5f * acc * (1.f + erff(acc * 0.70710678118f));
    out[(size_t)g * 768 + c] = (h16)ge;
  }
}

// ---------------------------------------------------------------------------
// Final fc2_B GEMM with fused bias + residual, f32 output:
// out[g] = x1[g] + t3[g] @ fc2_Bw + fc2_Bb.  grid = (M/32, 192/32).
// ---------------------------------------------------------------------------
__global__ __launch_bounds__(32)
void k_fc2b_res(const h16* __restrict__ A, const h16* __restrict__ Bt,
                const float* __restrict__ bias, const float* __restrict__ x1,
                float* __restrict__ out) {
  const int lane = threadIdx.x;
  const int m0 = blockIdx.x * 32, n0 = blockIdx.y * 32;
  v8f c00 = {}, c01 = {}, c10 = {}, c11 = {};
  gemm_tile_2x2<96>(A, Bt, m0, n0, c00, c01, c10, c11);
  const int col0 = n0 + (lane & 15), col1 = col0 + 16;
  const float bv0 = bias[col0], bv1 = bias[col1];
  const int mrow = m0 + ((lane >> 4) << 3);
#pragma unroll
  for (int rr = 0; rr < 8; ++rr) {
    size_t o0 = (size_t)(mrow + rr) * DIMC;
    size_t o1 = (size_t)(mrow + 16 + rr) * DIMC;
    out[o0 + col0] = x1[o0 + col0] + c00[rr] + bv0;
    out[o0 + col1] = x1[o0 + col1] + c01[rr] + bv1;
    out[o1 + col0] = x1[o1 + col0] + c10[rr] + bv0;
    out[o1 + col1] = x1[o1 + col1] + c11[rr] + bv1;
  }
}

// ---------------------------------------------------------------------------
// Host launcher
// ---------------------------------------------------------------------------
extern "C" void kernel_launch(void* const* d_in, const int* in_sizes, int n_in,
                              void* d_out, int out_size, void* d_ws, size_t ws_size,
                              hipStream_t stream) {
  (void)in_sizes; (void)n_in; (void)out_size; (void)ws_size;
  const float* x      = (const float*)d_in[0];
  const float* n1g    = (const float*)d_in[1];
  const float* n1b    = (const float*)d_in[2];
  const float* qkv_w  = (const float*)d_in[3];
  const float* qkv_b  = (const float*)d_in[4];
  const float* relb   = (const float*)d_in[5];
  const float* proj_w = (const float*)d_in[6];
  const float* proj_b = (const float*)d_in[7];
  const float* n2g    = (const float*)d_in[8];
  const float* n2b    = (const float*)d_in[9];
  const float* fc1A   = (const float*)d_in[10];
  const float* fc1Bw  = (const float*)d_in[11];
  const float* fc1Bb  = (const float*)d_in[12];
  const float* dww    = (const float*)d_in[13];
  const float* dwb    = (const float*)d_in[14];
  const float* fc2A   = (const float*)d_in[15];
  const float* fc2Bw  = (const float*)d_in[16];
  const float* fc2Bb  = (const float*)d_in[17];
  float* out = (float*)d_out;

  // Workspace layout (~253 MB), strict producer->consumer reuse of R1/R2.
  char* ws = (char*)d_ws;
  float* x1 = (float*)ws;                              // N*192 f32 (persistent)
  h16*   R1 = (h16*)(ws + 50331648);                   // 100 MB region
  h16*   R2 = (h16*)(ws + 150994944);                  // 100 MB region
  h16*   W  = (h16*)(ws + 251658240);                  // converted weights
  h16* qkv_wT  = W;                                    // 576 x 192
  h16* proj_wT = W + 110592;                           // 192 x 192
  h16* fc1_AT  = W + 147456;                           // 384 x 192
  h16* fc1_BwT = W + 221184;                           // 768 x 384
  h16* fc2_AT  = W + 516096;                           //  96 x 768
  h16* fc2_BwT = W + 589824;                           // 192 x  96
  float* qkv_bs = (float*)(ws + 251658240 + 608256 * 2); // 576 f32 (q cols scaled)

  h16* xw   = R1;  // windowed LN1 output, N x 192
  h16* qkvb = R2;  // N x 576 (q pre-scaled | k | v)
  h16* attn = R1;  // N x 192 (xw dead)
  h16* h2   = R1;  // N x 192 (attn dead after proj)
  h16* t1   = R2;  // N x 384 (qkv dead)
  h16* t2   = R1;  // N x 768 (h2 dead)
  h16* gbuf = R2;  // N x 768 (t1 dead)
  h16* t3   = R1;  // N x  96 (t2 dead)

  // --- weight prep (tiny; fold q-scale into qkv weight + bias) ---
  k_convert_wT<<<(192 * 576 + 255) / 256, 256, 0, stream>>>(qkv_w, qkv_wT, 192, 576, SCALE_Q, 192);
  k_convert_wT<<<(192 * 192 + 255) / 256, 256, 0, stream>>>(proj_w, proj_wT, 192, 192, 1.f, 0);
  k_convert_wT<<<(192 * 384 + 255) / 256, 256, 0, stream>>>(fc1A, fc1_AT, 192, 384, 1.f, 0);
  k_convert_wT<<<(384 * 768 + 255) / 256, 256, 0, stream>>>(fc1Bw, fc1_BwT, 384, 768, 1.f, 0);
  k_convert_wT<<<(768 * 96 + 255) / 256, 256, 0, stream>>>(fc2A, fc2_AT, 768, 96, 1.f, 0);
  k_convert_wT<<<(96 * 192 + 255) / 256, 256, 0, stream>>>(fc2Bw, fc2_BwT, 96, 192, 1.f, 0);
  k_scale_bias<<<3, 256, 0, stream>>>(qkv_b, qkv_bs, 576, 192, SCALE_Q);

  const int MT = N_TOK / 32;  // 2048 M tiles (32-row waves)

  // --- attention branch ---
  k_ln<<<N_TOK / 4, 128, 0, stream>>>(x, n1g, n1b, xw, 1);
  k_gemm_f16<192><<<dim3(MT, 576 / 32), 32, 0, stream>>>(xw, qkv_wT, qkv_bs, qkvb, 576);
  k_attn<<<256 * 6 * 16, 32, 0, stream>>>(qkvb, attn, relb);
  k_proj_res<<<dim3(MT, 192 / 32), 32, 0, stream>>>(attn, proj_wT, proj_b, x, x1);

  // --- MixFFN3D branch ---
  k_ln<<<N_TOK / 4, 128, 0, stream>>>(x1, n2g, n2b, h2, 0);
  k_gemm_f16<192><<<dim3(MT, 384 / 32), 32, 0, stream>>>(h2, fc1_AT, nullptr, t1, 384);
  k_gemm_f16<384><<<dim3(MT, 768 / 32), 32, 0, stream>>>(t1, fc1_BwT, fc1Bb, t2, 768);
  k_dwconv_gelu<<<N_TOK, 256, 0, stream>>>(t2, dww, dwb, gbuf);
  k_gemm_f16<768><<<dim3(MT, 96 / 32), 32, 0, stream>>>(gbuf, fc2_AT, nullptr, t3, 96);
  k_fc2b_res<<<dim3(MT, 192 / 32), 32, 0, stream>>>(t3, fc2_BwT, fc2Bb, x1, out);
}
